// ImageQualityAggregator_86638080294934
// MI455X (gfx1250) — matrix operand order repaired and checked by reading, not compiled, not run on hardware
//
#include <hip/hip_runtime.h>
#include <hip/hip_bf16.h>

// ---------- types ----------
typedef __attribute__((ext_vector_type(16))) __bf16 v16bf;
typedef __attribute__((ext_vector_type(8)))  float  v8f;
typedef __attribute__((ext_vector_type(4)))  unsigned short us4;
typedef __attribute__((address_space(3)))    unsigned char lds_byte;

#define NROW 9216
#define DDIM 1024
#define NTILE 72          // 9216 / 128
#define BM 128
#define BN 128
#define BK 32
#define SP 40             // LDS row stride in bf16 elems (padded: 80B, avoids bank conflicts)

static __device__ __forceinline__ unsigned short bf16_trunc(float x) {
  return (unsigned short)(__float_as_uint(x) >> 16);
}
static __device__ __forceinline__ float bf16_tof(unsigned short h) {
  return __uint_as_float(((unsigned int)h) << 16);
}
static __device__ __forceinline__ void split4(const float4 v, us4& hi, us4& lo) {
  float f[4] = {v.x, v.y, v.z, v.w};
#pragma unroll
  for (int i = 0; i < 4; ++i) {
    unsigned short h = bf16_trunc(f[i]);
    hi[i] = h;
    lo[i] = bf16_trunc(f[i] - bf16_tof(h));
  }
}
static __device__ __forceinline__ v16bf frag_ld(const unsigned short* p0,
                                                const unsigned short* p1) {
  union { v16bf v; uint4 q[2]; } u;
  u.q[0] = *(const uint4*)p0;   // -> ds_load_b128
  u.q[1] = *(const uint4*)p1;   // -> ds_load_b128
  return u.v;
}
static __device__ __forceinline__ v8f wmma_bf16(v16bf a, v16bf b, v8f c) {
  // D = A(16x32 bf16) * B(32x16 bf16) + C(f32)
  return __builtin_amdgcn_wmma_f32_16x16x32_bf16(false, a, false, b,
                                                 (short)0, c, false, false);
}

// CDNA5 async global->LDS copy (16B per lane), tracked by ASYNCcnt.
static __device__ __forceinline__ void async_g2l_b128(const float* gsrc, float* ldst) {
  unsigned lds_off = (unsigned)(unsigned long long)(lds_byte*)(void*)ldst;
  unsigned long long ga = (unsigned long long)gsrc;
  asm volatile("global_load_async_to_lds_b128 %0, %1, off"
               :: "v"(lds_off), "v"(ga) : "memory");
}
static __device__ __forceinline__ void wait_asynccnt0() {
#if __has_builtin(__builtin_amdgcn_s_wait_asynccnt)
  __builtin_amdgcn_s_wait_asynccnt(0);
#else
  asm volatile("s_wait_asynccnt 0" ::: "memory");
#endif
}

// ---------- kernel 0: zero accumulators ----------
__global__ void init_kernel(double* accum) {
  accum[0] = 0.0;  // sim sum
  accum[1] = 0.0;  // agg sum
}

// ---------- kernel 1: per-row squared norm + norm ----------
__global__ __launch_bounds__(256) void rowstats_kernel(const float* __restrict__ F,
                                                       float* __restrict__ sq,
                                                       float* __restrict__ nrm) {
  const int row = blockIdx.x;
  const int t = threadIdx.x;
  float4 v = *(const float4*)(F + (size_t)row * DDIM + t * 4);
  float p = v.x * v.x + v.y * v.y + v.z * v.z + v.w * v.w;
#pragma unroll
  for (int m = 16; m >= 1; m >>= 1) p += __shfl_xor(p, m, 32);
  __shared__ float s[8];
  if ((t & 31) == 0) s[t >> 5] = p;
  __syncthreads();
  if (t == 0) {
    float tot = 0.f;
#pragma unroll
    for (int i = 0; i < 8; ++i) tot += s[i];
    sq[row] = tot;
    nrm[row] = sqrtf(tot);
  }
}

// ---------- kernel 2: MLP  (x@W1 -> BN -> relu -> @W2 -> BN -> relu -> @W3 -> sigmoid) ----------
#define MROWS 8
#define XS 1028   // padded row stride (16B-aligned rows)
__global__ __launch_bounds__(128) void mlp_kernel(
    const float* __restrict__ F, const float* __restrict__ scores,
    const float* __restrict__ W1, const float* __restrict__ b1,
    const float* __restrict__ g1, const float* __restrict__ be1,
    const float* __restrict__ W2, const float* __restrict__ b2,
    const float* __restrict__ g2, const float* __restrict__ be2,
    const float* __restrict__ W3, const float* __restrict__ b3,
    double* __restrict__ accum) {
  __shared__ float xs[MROWS * XS];
  __shared__ float h1s[MROWS * 128];
  __shared__ float h2s[MROWS * 64];
  __shared__ float red[MROWS];
  const int t = threadIdx.x;
  const int r0 = blockIdx.x * MROWS;

  // stage MROWS feature rows via async global->LDS DMA (2048 float4, 128 thr -> 16 each)
#pragma unroll
  for (int it = 0; it < 16; ++it) {
    int idx = t + 128 * it;          // 0..2047
    int r = idx >> 8;                // 256 float4 per row
    int c4 = (idx & 255) << 2;
    async_g2l_b128(F + (size_t)(r0 + r) * DDIM + c4, &xs[r * XS + c4]);
  }
  if (t < MROWS) xs[t * XS + 1024] = scores[r0 + t];
  wait_asynccnt0();
  __syncthreads();

  const float inv_bn = rsqrtf(1.f + 1e-5f);

  // layer 1: each thread owns output column t (0..127) for all MROWS rows
  {
    float acc[MROWS];
#pragma unroll
    for (int r = 0; r < MROWS; ++r) acc[r] = 0.f;
    for (int k = 0; k < 1024; ++k) {
      float w = W1[k * 128 + t];
#pragma unroll
      for (int r = 0; r < MROWS; ++r) acc[r] += xs[r * XS + k] * w;
    }
    {
      float w = W1[1024 * 128 + t];
#pragma unroll
      for (int r = 0; r < MROWS; ++r) acc[r] += xs[r * XS + 1024] * w;
    }
    float bb = b1[t], gg = g1[t], be = be1[t];
#pragma unroll
    for (int r = 0; r < MROWS; ++r) {
      float h = gg * ((acc[r] + bb) * inv_bn) + be;
      h1s[r * 128 + t] = fmaxf(h, 0.f);
    }
  }
  __syncthreads();

  // layer 2: threads 0..63 own output column t
  if (t < 64) {
    float acc[MROWS];
#pragma unroll
    for (int r = 0; r < MROWS; ++r) acc[r] = 0.f;
    for (int k = 0; k < 128; ++k) {
      float w = W2[k * 64 + t];
#pragma unroll
      for (int r = 0; r < MROWS; ++r) acc[r] += h1s[r * 128 + k] * w;
    }
    float bb = b2[t], gg = g2[t], be = be2[t];
#pragma unroll
    for (int r = 0; r < MROWS; ++r) {
      float h = gg * ((acc[r] + bb) * inv_bn) + be;
      h2s[r * 64 + t] = fmaxf(h, 0.f);
    }
  }
  __syncthreads();

  // layer 3 + sigmoid: threads 0..MROWS-1 own one row
  if (t < MROWS) {
    float a = b3[0];
    for (int k = 0; k < 64; ++k) a += h2s[t * 64 + k] * W3[k];
    red[t] = 1.f / (1.f + __expf(-a));
  }
  __syncthreads();
  if (t == 0) {
    float s = 0.f;
#pragma unroll
    for (int i = 0; i < MROWS; ++i) s += red[i];
    atomicAdd(accum + 1, (double)s);
  }
}

// ---------- kernel 3: Gram via split-bf16 WMMA, fused sim transform + reduce ----------
__global__ __launch_bounds__(256) void sim_gemm_kernel(const float* __restrict__ F,
                                                       const float* __restrict__ sq,
                                                       const float* __restrict__ nrm,
                                                       double* __restrict__ accum) {
  __shared__ unsigned short sAhi[BM * SP], sAlo[BM * SP];
  __shared__ unsigned short sBhi[BN * SP], sBlo[BN * SP];
  __shared__ float s_sqi[BM], s_ni[BM], s_sqj[BN], s_nj[BN];
  __shared__ float sred[8];

  // triangular tile decode (block-uniform scalar loop)
  int ib = 0, rem = blockIdx.x;
  while (rem >= (NTILE - ib)) { rem -= (NTILE - ib); ++ib; }
  const int jb = ib + rem;
  const int i0 = ib * BM, j0 = jb * BN;

  const int t = threadIdx.x;
  const int lane = t & 31, wave = t >> 5;
  const int wm = wave >> 2;      // 0..1 : 64-row band
  const int wn = wave & 3;       // 0..3 : 32-col band
  const int lr = lane & 15, lh = lane >> 4;

  if (t < 128) { s_sqi[t] = sq[i0 + t]; s_ni[t] = nrm[i0 + t]; }
  else { int u = t - 128; s_sqj[u] = sq[j0 + u]; s_nj[u] = nrm[j0 + u]; }

  // register double-buffer for the staged tiles (8 x float4 per thread)
  float4 ra[4], rb[4];
  auto load_tile = [&](int k0) {
#pragma unroll
    for (int it = 0; it < 4; ++it) {
      int idx = t + 256 * it;        // 0..1023
      int row = idx >> 3;            // 0..127
      int c = (idx & 7) << 2;        // 0,4,...,28
      const float* pa = F + (size_t)(i0 + row) * DDIM + k0 + c;
      const float* pb = F + (size_t)(j0 + row) * DDIM + k0 + c;
      ra[it] = *(const float4*)pa;
      rb[it] = *(const float4*)pb;
      if (k0 + BK < DDIM) {          // warm L2 for the tile after next
        __builtin_prefetch(pa + BK, 0, 1);
        __builtin_prefetch(pb + BK, 0, 1);
      }
    }
  };
  auto store_tile = [&]() {
#pragma unroll
    for (int it = 0; it < 4; ++it) {
      int idx = t + 256 * it;
      int row = idx >> 3;
      int c = (idx & 7) << 2;
      us4 hi, lo;
      split4(ra[it], hi, lo);
      *(us4*)&sAhi[row * SP + c] = hi;
      *(us4*)&sAlo[row * SP + c] = lo;
      split4(rb[it], hi, lo);
      *(us4*)&sBhi[row * SP + c] = hi;
      *(us4*)&sBlo[row * SP + c] = lo;
    }
  };

  v8f acc[4][2] = {};
  load_tile(0);

  for (int k0 = 0; k0 < DDIM; k0 += BK) {
    __syncthreads();                 // LDS free (previous compute done)
    store_tile();
    __syncthreads();
    if (k0 + BK < DDIM) load_tile(k0 + BK);   // overlap next-tile loads with WMMAs

    // fragment loads per ISA 16-bit layouts (A 16x32, B 32x16; wave32)
    v16bf ahi[4], alo[4], bhi[2], blo[2];
    const int kb = lh * 8;
#pragma unroll
    for (int mi = 0; mi < 4; ++mi) {
      int r = (wm * 64 + mi * 16 + lr) * SP;
      ahi[mi] = frag_ld(&sAhi[r + kb], &sAhi[r + 16 + kb]);
      alo[mi] = frag_ld(&sAlo[r + kb], &sAlo[r + 16 + kb]);
    }
#pragma unroll
    for (int ni = 0; ni < 2; ++ni) {
      int r = (wn * 32 + ni * 16 + lr) * SP + lh * 8;
      bhi[ni] = frag_ld(&sBhi[r], &sBhi[r + 16]);
      blo[ni] = frag_ld(&sBlo[r], &sBlo[r + 16]);
    }

    // split-bf16 product: hi*hi + lo*hi + hi*lo  (24 WMMAs / K-step)
#pragma unroll
    for (int mi = 0; mi < 4; ++mi)
#pragma unroll
      for (int ni = 0; ni < 2; ++ni) {
        acc[mi][ni] = wmma_bf16(ahi[mi], bhi[ni], acc[mi][ni]);
        acc[mi][ni] = wmma_bf16(alo[mi], bhi[ni], acc[mi][ni]);
        acc[mi][ni] = wmma_bf16(ahi[mi], blo[ni], acc[mi][ni]);
      }
  }

  // fused epilogue: gaussian similarity + partial sum
  float lsum = 0.f;
#pragma unroll
  for (int mi = 0; mi < 4; ++mi)
#pragma unroll
    for (int ni = 0; ni < 2; ++ni) {
      const int jc = wn * 32 + ni * 16 + lr;          // C layout: N = lane&15
      const float sqj = s_sqj[jc], nj = s_nj[jc];
      const int ir0 = wm * 64 + mi * 16 + lh * 8;     // M = vgpr + (lane>>4)*8
#pragma unroll
      for (int v = 0; v < 8; ++v) {
        const int il = ir0 + v;
        float g = acc[mi][ni][v];
        float cosv = g / fmaxf(s_ni[il] * nj, 1e-6f);
        float d2 = fmaxf(s_sqi[il] + sqj - 2.f * g, 0.f);
        float den = fmaxf(1024.f * (1.f + cosv), 1e-6f);
        float s = __expf(-d2 / den);
        if (i0 + il == j0 + jc) s = 1.f;              // exact diagonal
        lsum += s;
      }
    }
  lsum *= (ib == jb) ? 1.f : 2.f;                     // symmetry weight

#pragma unroll
  for (int m = 16; m >= 1; m >>= 1) lsum += __shfl_xor(lsum, m, 32);
  if (lane == 0) sred[wave] = lsum;
  __syncthreads();
  if (t == 0) {
    float tot = 0.f;
#pragma unroll
    for (int i = 0; i < 8; ++i) tot += sred[i];
    atomicAdd(accum, (double)tot);
  }
}

// ---------- kernel 4: scalar combine ----------
__global__ void finalize_kernel(const double* __restrict__ accum, float* __restrict__ out) {
  const double n = (double)NROW;
  out[0] = (float)((accum[1] / n) * (accum[0] / (n * n)));
}

// ---------- launch ----------
extern "C" void kernel_launch(void* const* d_in, const int* in_sizes, int n_in,
                              void* d_out, int out_size, void* d_ws, size_t ws_size,
                              hipStream_t stream) {
  (void)in_sizes; (void)n_in; (void)out_size; (void)ws_size;
  const float* F      = (const float*)d_in[0];
  const float* scores = (const float*)d_in[1];
  const float* W1  = (const float*)d_in[2];
  const float* b1  = (const float*)d_in[3];
  const float* g1  = (const float*)d_in[4];
  const float* be1 = (const float*)d_in[5];
  const float* W2  = (const float*)d_in[6];
  const float* b2  = (const float*)d_in[7];
  const float* g2  = (const float*)d_in[8];
  const float* be2 = (const float*)d_in[9];
  const float* W3  = (const float*)d_in[10];
  const float* b3  = (const float*)d_in[11];

  float*  sq    = (float*)d_ws;
  float*  nrm   = sq + NROW;
  double* accum = (double*)((char*)d_ws + (size_t)2 * NROW * sizeof(float)); // 8B aligned

  init_kernel<<<1, 1, 0, stream>>>(accum);
  rowstats_kernel<<<NROW, 256, 0, stream>>>(F, sq, nrm);
  mlp_kernel<<<NROW / MROWS, 128, 0, stream>>>(F, scores, W1, b1, g1, be1,
                                               W2, b2, g2, be2, W3, b3, accum);
  sim_gemm_kernel<<<NTILE * (NTILE + 1) / 2, 256, 0, stream>>>(F, sq, nrm, accum);
  finalize_kernel<<<1, 1, 0, stream>>>(accum, (float*)d_out);
}
